// EqvLBAFeedForward_51994874085873
// MI455X (gfx1250) — compile-verified, matrix operand order checked by reading, not run on hardware
//
#include <hip/hip_runtime.h>
#include <hip/hip_bf16.h>
#include <math.h>

typedef __attribute__((ext_vector_type(16))) _Float16 v16h;
typedef __attribute__((ext_vector_type(8)))  _Float16 v8h;
typedef __attribute__((ext_vector_type(8)))  float    v8f;
typedef __attribute__((ext_vector_type(8)))  float    v8fv;

#define BZ 4     // batch
#define NP 256   // points
#define CC 32    // channels
#define HH 100   // radial hidden
#define HP 128   // padded hidden (K granularity 32)

// ws layout: Wbuf = BZ*NP*CC*HP f16 (8.39 MB, [z,b][i][h] h-innermost),
//            then ppart = 16*BZ*CC f32 (8 KB)
#define W_ELEMS (BZ * NP * CC * HP)

// ---------------------------------------------------------------------------
// Kernel 1: W[z,b][i][h] = sum_j rw2[h, i*C + j] * x[z,b,j], stored f16 with
// h innermost and padded to 128 -> B fragments become contiguous 16B runs.
// ---------------------------------------------------------------------------
__global__ __launch_bounds__(256)
void k1_compute_W(const float* __restrict__ x, const float* __restrict__ rw2,
                  _Float16* __restrict__ Wbuf) {
  const int zb  = blockIdx.x;          // 0 .. BZ*NP-1
  const int tid = threadIdx.x;
  __shared__ float xs[CC];
  if (tid < CC) xs[tid] = x[zb * CC + tid];
  __syncthreads();
  for (int o = tid; o < CC * HP; o += 256) {
    const int i = o >> 7, h = o & (HP - 1);
    float acc = 0.0f;
    if (h < HH) {
      const float* row = rw2 + (h * (CC * CC) + i * CC);
#pragma unroll
      for (int j = 0; j < CC; ++j) acc += row[j] * xs[j];
    }
    Wbuf[(size_t)zb * (CC * HP) + o] = (_Float16)acc;
  }
}

// ---------------------------------------------------------------------------
// Kernel 2: fused point-conv. Grid (16 a-tiles, BZ). 4 waves/block, each wave
// strides over b; hid generated in-register straight into WMMA A-fragment
// layout (vector LDS reads of rw1); B-fragments are 2x global_load_b128 each.
// ---------------------------------------------------------------------------
__global__ __launch_bounds__(128)
void k2_conv_pool(const float* __restrict__ xyz, const int* __restrict__ mask,
                  const float* __restrict__ rw1, const _Float16* __restrict__ Wbuf,
                  float* __restrict__ ppart) {
  const int z     = blockIdx.y;
  const int atile = blockIdx.x;
  const int tid   = threadIdx.x;
  const int wave  = tid >> 5;
  const int lane  = tid & 31;

  __shared__ __align__(32) float lrw1[3 * HP];
  __shared__ __align__(16) float red[4][16][CC];

  for (int o = tid; o < 3 * HP; o += 128) {
    const int k = o / HP, h = o % HP;
    lrw1[o] = (h < HH) ? rw1[k * HH + h] : 0.0f;   // zero-pad -> swish(0)=0
  }
  __syncthreads();

  const int m     = lane & 15;               // M row (A) / N col (B) for this lane
  const int a     = atile * 16 + m;
  const int baseK = (lane < 16) ? 0 : 8;     // 16-bit fragment K layout

  const float ax = xyz[(z * NP + a) * 3 + 0];
  const float ay = xyz[(z * NP + a) * 3 + 1];
  const float az = xyz[(z * NP + a) * 3 + 2];

  v8f acc0 = {};
  v8f acc1 = {};

  for (int b = wave; b < NP; b += 4) {
    const float bx = xyz[(z * NP + b) * 3 + 0];
    const float by = xyz[(z * NP + b) * 3 + 1];
    const float bz = xyz[(z * NP + b) * 3 + 2];
    const float dx = ax - bx, dy = ay - by, dz = az - bz;
    const float r  = sqrtf(dx * dx + dy * dy + dz * dz + 1e-12f);
    // centers {0,5,10}, gamma = NUM_BASIS / MAX_RADIUS = 0.3
    const float t0 = r, t1 = r - 5.0f, t2 = r - 10.0f;
    const float g0 = __expf(-0.3f * t0 * t0);
    const float g1 = __expf(-0.3f * t1 * t1);
    const float g2 = __expf(-0.3f * t2 * t2);

    const _Float16* wr = Wbuf + (size_t)(z * NP + b) * (CC * HP);
    if (b + 4 < NP) {
      // Pull next iteration's W row toward the WGP (global_prefetch_b8 path)
      __builtin_prefetch(wr + (size_t)4 * (CC * HP) + m * HP, 0, 1);
    }

#pragma unroll
    for (int c = 0; c < 4; ++c) {
      const int kb = c * 32 + baseK;

      // ---- A fragment: hid[a, h]; elems 0..7 <-> K=kb+e, 8..15 <-> K=kb+16+e
      v16h afrag = {};
#pragma unroll
      for (int run = 0; run < 2; ++run) {
        const int hb = kb + run * 16;
        const v8fv w0 = *(const v8fv*)&lrw1[hb];            // 32B LDS vector loads
        const v8fv w1 = *(const v8fv*)&lrw1[HP + hb];
        const v8fv w2 = *(const v8fv*)&lrw1[2 * HP + hb];
        const v8fv sv = g0 * w0 + g1 * w1 + g2 * w2;
#pragma unroll
        for (int e = 0; e < 8; ++e) {
          const float s   = sv[e];
          // swish via fast hardware reciprocal (result is truncated to f16,
          // so 1-ulp v_rcp_f32 is exact enough; avoids IEEE div expansion)
          const float hid = s * __builtin_amdgcn_rcpf(1.0f + __expf(-s));
          afrag[run * 8 + e] = (_Float16)hid;
        }
      }

      // ---- B fragments: contiguous 8-half runs -> global_load_b128 each
      const v8h* pb0 = (const v8h*)(wr + m * HP + kb);            // i = m
      const v8h* pb1 = (const v8h*)(wr + (16 + m) * HP + kb);     // i = 16+m
      const v8h b0lo = pb0[0], b0hi = pb0[2];                     // K=kb..+7, kb+16..+23
      const v8h b1lo = pb1[0], b1hi = pb1[2];
      const v16h bf0 = __builtin_shufflevector(b0lo, b0hi,
          0, 1, 2, 3, 4, 5, 6, 7, 8, 9, 10, 11, 12, 13, 14, 15);
      const v16h bf1 = __builtin_shufflevector(b1lo, b1hi,
          0, 1, 2, 3, 4, 5, 6, 7, 8, 9, 10, 11, 12, 13, 14, 15);

      acc0 = __builtin_amdgcn_wmma_f32_16x16x32_f16(false, afrag, false, bf0,
                                                    (short)0, acc0, false, false);
      acc1 = __builtin_amdgcn_wmma_f32_16x16x32_f16(false, afrag, false, bf1,
                                                    (short)0, acc1, false, false);
    }
  }

  // Spill per-wave partial 16x32 tiles (C/D layout: lanes0-15 M=v, lanes16-31 M=v+8)
#pragma unroll
  for (int v = 0; v < 8; ++v) {
    const int mm = (lane < 16) ? v : (v + 8);
    red[wave][mm][m]      = acc0[v];
    red[wave][mm][16 + m] = acc1[v];
  }
  __syncthreads();

  // Cross-wave sum, scale 1/sqrt(N), abs, mask, partial pool over this a-tile
  const int i  = tid & 31;
  const int mb = tid >> 5;
  float local = 0.0f;
#pragma unroll
  for (int q = 0; q < 4; ++q) {
    const int mm = mb + 4 * q;
    float s = red[0][mm][i] + red[1][mm][i] + red[2][mm][i] + red[3][mm][i];
    s = fabsf(s) * (1.0f / 16.0f);                 // / sqrt(256)
    const int ag = atile * 16 + mm;
    if (mask[z * NP + ag] != 0) local += s;
  }
  __syncthreads();
  red[0][mb][i] = local;
  __syncthreads();
  if (tid < 32) {
    const float p = red[0][0][tid] + red[0][1][tid] + red[0][2][tid] + red[0][3][tid];
    ppart[(atile * BZ + z) * CC + tid] = p;        // deterministic partials
  }
}

// ---------------------------------------------------------------------------
// Kernel 3: pool partials -> normalize (ddof=1) -> leaky(fc3) -> fc2
// One wave per batch element.
// ---------------------------------------------------------------------------
__global__ __launch_bounds__(128)
void k3_head(const float* __restrict__ ppart, const float* __restrict__ fc3_w,
             const float* __restrict__ fc3_b, const float* __restrict__ fc2_w,
             const float* __restrict__ fc2_b, float* __restrict__ out) {
  const int z    = threadIdx.x >> 5;
  const int lane = threadIdx.x & 31;
  __shared__ float nb[BZ][CC];

  float v = 0.0f;
#pragma unroll
  for (int t = 0; t < 16; ++t) v += ppart[(t * BZ + z) * CC + lane];

  float s = v;
  for (int off = 16; off > 0; off >>= 1) s += __shfl_xor(s, off, 32);
  const float mean = s * (1.0f / CC);
  const float d = v - mean;
  float q = d * d;
  for (int off = 16; off > 0; off >>= 1) q += __shfl_xor(q, off, 32);
  const float stdv = sqrtf(q * (1.0f / (CC - 1)));     // unbiased (ddof=1)
  nb[z][lane] = d / (stdv + 1e-6f);                    // keep IEEE div here
  __syncthreads();

  float h1 = fc3_b[lane];
#pragma unroll
  for (int j = 0; j < CC; ++j) h1 += nb[z][j] * fc3_w[j * CC + lane];
  h1 = (h1 >= 0.0f) ? h1 : 0.01f * h1;

  float c = h1 * fc2_w[lane];
  for (int off = 16; off > 0; off >>= 1) c += __shfl_xor(c, off, 32);
  if (lane == 0) out[z] = c + fc2_b[0];
}

// ---------------------------------------------------------------------------
extern "C" void kernel_launch(void* const* d_in, const int* in_sizes, int n_in,
                              void* d_out, int out_size, void* d_ws, size_t ws_size,
                              hipStream_t stream) {
  const float* x     = (const float*)d_in[0];
  const float* xyz   = (const float*)d_in[1];
  const int*   mask  = (const int*)d_in[2];
  const float* rw1   = (const float*)d_in[3];
  const float* rw2   = (const float*)d_in[4];
  const float* fc3_w = (const float*)d_in[5];
  const float* fc3_b = (const float*)d_in[6];
  const float* fc2_w = (const float*)d_in[7];
  const float* fc2_b = (const float*)d_in[8];

  _Float16* Wbuf  = (_Float16*)d_ws;                       // 8.39 MB (L2-resident)
  float*    ppart = (float*)((char*)d_ws + (size_t)W_ELEMS * sizeof(_Float16));

  k1_compute_W<<<BZ * NP, 256, 0, stream>>>(x, rw2, Wbuf);
  k2_conv_pool<<<dim3(16, BZ), 128, 0, stream>>>(xyz, mask, rw1, Wbuf, ppart);
  k3_head<<<1, 128, 0, stream>>>(ppart, fc3_w, fc3_b, fc2_w, fc2_b, (float*)d_out);
}
